// AxialTL_21002390077470
// MI455X (gfx1250) — matrix operand order repaired and checked by reading, not compiled, not run on hardware
//
#include <hip/hip_runtime.h>
#include <math.h>

// ---------------- problem constants ----------------
constexpr int R_ = 64, C_ = 256, E_ = 768, H_ = 8, DK_ = 96, F_ = 3072;
constexpr int M_ = R_ * C_;          // 16384 tokens (B == 1)
constexpr int LDT = 40;              // LDS row stride (32 + 8 pad), 80B -> 16B aligned

// ---------------- WMMA types ----------------
typedef __attribute__((ext_vector_type(16))) __bf16 v16bf;
typedef __attribute__((ext_vector_type(8)))  float  v8f;

union FragU { v16bf v; uint4 q[2]; };

__device__ inline v8f wmma_bf16(v16bf a, v16bf b, v8f c) {
  return __builtin_amdgcn_wmma_f32_16x16x32_bf16(false, a, false, b, (short)0, c,
                                                 false, false);
}

// CDNA5 async copy: global -> LDS, 16B per lane, tracked by ASYNCcnt.
// INST_OFFSET is added to BOTH the LDS and global address (ISA 10.x async pseudocode),
// so the offset:16 form covers the second half of a 32B stage with no address math.
__device__ inline void async_ld_b128(unsigned lds_byte_addr, const void* gptr) {
  asm volatile("global_load_async_to_lds_b128 %0, %1, off"
               :
               : "v"(lds_byte_addr), "v"(gptr)
               : "memory");
}
__device__ inline void async_ld_b128_o16(unsigned lds_byte_addr, const void* gptr) {
  asm volatile("global_load_async_to_lds_b128 %0, %1, off offset:16"
               :
               : "v"(lds_byte_addr), "v"(gptr)
               : "memory");
}
__device__ inline void async_ld_32B(unsigned lds_byte_addr, const void* gptr) {
  async_ld_b128(lds_byte_addr, gptr);
  async_ld_b128_o16(lds_byte_addr, gptr);
}
__device__ inline void wait_async0() {
  asm volatile("s_wait_asynccnt 0x0" ::: "memory");
}
__device__ inline unsigned lds_addr(const void* p) {   // low 32 bits = DS offset
  return (unsigned)(size_t)p;
}

// A-matrix 16x32 bf16 fragment (ISA 7.12.2): lane L<16: row L, K {0..7,16..23};
// lane L>=16: row L-16, K {8..15,24..31}. Two b128 LDS loads.
__device__ inline v16bf ld_frag_a(const unsigned short* lds, int rowBase) {
  const int lane = (int)(threadIdx.x & 31u);
  const unsigned short* p = lds + (size_t)(rowBase + (lane & 15)) * LDT + ((lane >> 4) << 3);
  FragU f;
  f.q[0] = *(const uint4*)(p);
  f.q[1] = *(const uint4*)(p + 16);
  return f.v;
}

// B-matrix 32x16 bf16 fragment: lane L: col N = L&15, K {0..7,16..23} (L<16) / {8..15,24..31} (L>=16).
// LDS tile stored [n][k].
__device__ inline v16bf ld_frag_b(const unsigned short* lds, int nBase) {
  const int lane = (int)(threadIdx.x & 31u);
  const unsigned short* p = lds + (size_t)(nBase + (lane & 15)) * LDT + ((lane >> 4) << 3);
  FragU f;
  f.q[0] = *(const uint4*)(p);
  f.q[1] = *(const uint4*)(p + 16);
  return f.v;
}

__device__ inline unsigned short f2bf(float f) {
  unsigned int u = __float_as_uint(f);
  unsigned int r = u + 0x7FFFu + ((u >> 16) & 1u);   // round-to-nearest-even
  return (unsigned short)(r >> 16);
}

__device__ inline float gelu_exact(float x) {
  return 0.5f * x * (1.0f + erff(x * 0.70710678118654752f));
}

// ---------------- f32 -> bf16 weight cast ----------------
__global__ __launch_bounds__(256) void k_cast(const float* __restrict__ s,
                                              unsigned short* __restrict__ d, int n) {
  int i = (int)(blockIdx.x * 256 + threadIdx.x) * 4;
  if (i + 3 < n) {
    float4 v = *(const float4*)(s + i);
    d[i] = f2bf(v.x); d[i + 1] = f2bf(v.y); d[i + 2] = f2bf(v.z); d[i + 3] = f2bf(v.w);
  } else {
    for (int j = i; j < n; ++j) d[j] = f2bf(s[j]);
  }
}

// ---------------- LayerNorm (f32 in, bf16 out), one 256-thread block per token ----
__global__ __launch_bounds__(256) void k_ln(const float* __restrict__ X,
                                            const float* __restrict__ w,
                                            const float* __restrict__ b,
                                            unsigned short* __restrict__ out) {
  const int t = blockIdx.x;
  const int tid = threadIdx.x;
  const float* x = X + (size_t)t * E_;
  float v[3], s = 0.f, ss = 0.f;
  for (int i = 0; i < 3; ++i) {
    v[i] = x[tid + i * 256];
    s += v[i]; ss += v[i] * v[i];
  }
  __shared__ float r1[256], r2[256];
  r1[tid] = s; r2[tid] = ss;
  __syncthreads();
  for (int st = 128; st > 0; st >>= 1) {
    if (tid < st) { r1[tid] += r1[tid + st]; r2[tid] += r2[tid + st]; }
    __syncthreads();
  }
  const float mean = r1[0] * (1.0f / E_);
  const float var  = r2[0] * (1.0f / E_) - mean * mean;
  const float rstd = rsqrtf(var + 1e-5f);
  for (int i = 0; i < 3; ++i) {
    int c = tid + i * 256;
    out[(size_t)t * E_ + c] = f2bf((v[i] - mean) * rstd * w[c] + b[c]);
  }
}

// ---------------- generic bf16 GEMM: Y[M,N] = act((A[M,K] @ W[N,K]^T + bias)*scale)+resid
// 256 thr / 8 waves, 128x128 tile, async double-buffered LDS staging (1 barrier/step)
__global__ __launch_bounds__(256) void k_gemm(const unsigned short* __restrict__ A,
                                              const unsigned short* __restrict__ W,
                                              const float* __restrict__ bias,
                                              const float* __restrict__ resid,
                                              float* __restrict__ outF,
                                              unsigned short* __restrict__ outB,
                                              int Md, int Nd, int Kd,
                                              float scale, int act) {
  __shared__ unsigned short lA[2][128 * LDT];
  __shared__ unsigned short lB[2][128 * LDT];
  const int tid = threadIdx.x, lane = tid & 31, wid = tid >> 5;
  const int wm = (wid & 3) * 32, wn = (wid >> 2) * 64;
  const int m0 = blockIdx.x * 128, n0 = blockIdx.y * 128;
  const int srow = tid >> 1, skp = (tid & 1) * 16;

  v8f acc[2][4];
  const v8f zero = {0.f,0.f,0.f,0.f,0.f,0.f,0.f,0.f};
  for (int i = 0; i < 2; ++i) for (int j = 0; j < 4; ++j) acc[i][j] = zero;

  const size_t arow = (size_t)(m0 + srow) * Kd + skp;
  const size_t brow = (size_t)(n0 + srow) * Kd + skp;
  unsigned la[2], lb[2];
  la[0] = lds_addr(&lA[0][srow * LDT + skp]);
  la[1] = lds_addr(&lA[1][srow * LDT + skp]);
  lb[0] = lds_addr(&lB[0][srow * LDT + skp]);
  lb[1] = lds_addr(&lB[1][srow * LDT + skp]);

  // prologue: tile 0 -> buffer 0
  async_ld_32B(la[0], A + arow);
  async_ld_32B(lb[0], W + brow);

  const int nk = Kd >> 5;
  for (int t = 0; t < nk; ++t) {
    wait_async0();
    __syncthreads();                       // tile t resident; all waves past compute t-1
    const int p = t & 1;
    if (t + 1 < nk) {
      const int q = p ^ 1;
      const size_t ko = (size_t)(t + 1) << 5;
      async_ld_32B(la[q], A + arow + ko);
      async_ld_32B(lb[q], W + brow + ko);
    }
    v16bf af0 = ld_frag_a(lA[p], wm);
    v16bf af1 = ld_frag_a(lA[p], wm + 16);
    for (int tn = 0; tn < 4; ++tn) {
      v16bf bf = ld_frag_b(lB[p], wn + tn * 16);
      acc[0][tn] = wmma_bf16(af0, bf, acc[0][tn]);
      acc[1][tn] = wmma_bf16(af1, bf, acc[1][tn]);
    }
  }

  const int er = (lane >> 4) << 3, ec = lane & 15;
  for (int tm = 0; tm < 2; ++tm)
    for (int tn = 0; tn < 4; ++tn)
      for (int i = 0; i < 8; ++i) {
        int row = m0 + wm + tm * 16 + er + i;
        int col = n0 + wn + tn * 16 + ec;
        float t = acc[tm][tn][i];
        if (bias) t += bias[col];
        t *= scale;
        if (act) t = gelu_exact(t);
        size_t idx = (size_t)row * Nd + col;
        if (resid) t += resid[idx];
        if (outF) outF[idx] = t;
        if (outB) outB[idx] = f2bf(t);
      }
}

// ---------------- row-attention logits: L[h,i,j] = sum_{r,d} q[r,i,h,d] k[r,j,h,d]
// per head: 256x256 GEMM, K = 64*96 = 6144 (32-chunks never cross head dim: 96 = 3*32)
__global__ __launch_bounds__(256) void k_row_logits(const unsigned short* __restrict__ Q,
                                                    const unsigned short* __restrict__ Km,
                                                    float* __restrict__ L) {
  __shared__ unsigned short lA[2][128 * LDT];
  __shared__ unsigned short lB[2][128 * LDT];
  const int tid = threadIdx.x, lane = tid & 31, wid = tid >> 5;
  const int wm = (wid & 3) * 32, wn = (wid >> 2) * 64;
  const int i0 = blockIdx.x * 128, j0 = blockIdx.y * 128, h = blockIdx.z;
  const int srow = tid >> 1, skp = (tid & 1) * 16;

  v8f acc[2][4];
  const v8f zero = {0.f,0.f,0.f,0.f,0.f,0.f,0.f,0.f};
  for (int i = 0; i < 2; ++i) for (int j = 0; j < 4; ++j) acc[i][j] = zero;

  unsigned la[2], lb[2];
  la[0] = lds_addr(&lA[0][srow * LDT + skp]);
  la[1] = lds_addr(&lA[1][srow * LDT + skp]);
  lb[0] = lds_addr(&lB[0][srow * LDT + skp]);
  lb[1] = lds_addr(&lB[1][srow * LDT + skp]);

  auto aoff = [&](int t) -> size_t {
    int r = t / 3, dc = t - r * 3;
    return (size_t)(r * C_ + i0 + srow) * E_ + (size_t)(h * DK_ + dc * 32 + skp);
  };
  auto boff = [&](int t) -> size_t {
    int r = t / 3, dc = t - r * 3;
    return (size_t)(r * C_ + j0 + srow) * E_ + (size_t)(h * DK_ + dc * 32 + skp);
  };

  const int nk = R_ * 3;
  async_ld_32B(la[0], Q + aoff(0));
  async_ld_32B(lb[0], Km + boff(0));

  for (int t = 0; t < nk; ++t) {
    wait_async0();
    __syncthreads();
    const int p = t & 1;
    if (t + 1 < nk) {
      const int q = p ^ 1;
      async_ld_32B(la[q], Q + aoff(t + 1));
      async_ld_32B(lb[q], Km + boff(t + 1));
    }
    v16bf af0 = ld_frag_a(lA[p], wm);
    v16bf af1 = ld_frag_a(lA[p], wm + 16);
    for (int tn = 0; tn < 4; ++tn) {
      v16bf bf = ld_frag_b(lB[p], wn + tn * 16);
      acc[0][tn] = wmma_bf16(af0, bf, acc[0][tn]);
      acc[1][tn] = wmma_bf16(af1, bf, acc[1][tn]);
    }
  }

  const int er = (lane >> 4) << 3, ec = lane & 15;
  for (int tm = 0; tm < 2; ++tm)
    for (int tn = 0; tn < 4; ++tn)
      for (int i = 0; i < 8; ++i) {
        int row = i0 + wm + tm * 16 + er + i;
        int col = j0 + wn + tn * 16 + ec;
        L[(size_t)h * C_ * C_ + (size_t)row * C_ + col] = acc[tm][tn][i];
      }
}

// ---------------- row softmax over 256, writes f32 probs (d_out) + bf16 copy ----
__global__ __launch_bounds__(256) void k_row_softmax(const float* __restrict__ L,
                                                     float* __restrict__ P,
                                                     unsigned short* __restrict__ Pb) {
  const int row = blockIdx.x;        // h*C + i
  const int tid = threadIdx.x;
  __shared__ float red[256];
  float v = L[(size_t)row * 256 + tid];
  red[tid] = v; __syncthreads();
  for (int s = 128; s > 0; s >>= 1) {
    if (tid < s) red[tid] = fmaxf(red[tid], red[tid + s]);
    __syncthreads();
  }
  const float mx = red[0]; __syncthreads();
  float e = __expf(v - mx);
  red[tid] = e; __syncthreads();
  for (int s = 128; s > 0; s >>= 1) {
    if (tid < s) red[tid] += red[tid + s];
    __syncthreads();
  }
  float p = e / red[0];
  P[(size_t)row * 256 + tid]  = p;
  Pb[(size_t)row * 256 + tid] = f2bf(p);
}

// ---------------- row PV: rc[r,i,h,d] = sum_j P[h,i,j] v[r,j,h,d]
// per (h,r): 256(i) x 96(d) x 256(j) GEMM; waves 4(M) x 2(N=48), 2x3 frags
__global__ __launch_bounds__(256) void k_row_pv(const unsigned short* __restrict__ P,
                                                const unsigned short* __restrict__ V,
                                                unsigned short* __restrict__ O) {
  __shared__ unsigned short lA[128 * LDT];
  __shared__ unsigned short lB[96 * LDT];
  const int tid = threadIdx.x, lane = tid & 31, wid = tid >> 5;
  const int wm = (wid & 3) * 32, wn = (wid >> 2) * 48;
  const int i0 = blockIdx.x * 128, r = blockIdx.y, h = blockIdx.z;
  const int srow = tid >> 1, skp = (tid & 1) * 16;
  const unsigned laddr = lds_addr(&lA[srow * LDT + skp]);

  v8f acc[2][3];
  const v8f zero = {0.f,0.f,0.f,0.f,0.f,0.f,0.f,0.f};
  for (int i = 0; i < 2; ++i) for (int j = 0; j < 3; ++j) acc[i][j] = zero;

  for (int k0 = 0; k0 < C_; k0 += 32) {
    __syncthreads();                 // all waves done reading previous tile
    const size_t pa = (size_t)h * C_ * C_ + (size_t)(i0 + srow) * C_ + k0 + skp;
    async_ld_32B(laddr, P + pa);     // A tile via async copy (no VGPR bounce)
    for (int it = 0; it < 12; ++it) {               // B tile 32(k=j) x 96(n=d) -> [n][k]
      int idx = tid * 12 + it;
      int kk = idx / 96, d = idx % 96;
      lB[d * LDT + kk] = V[(size_t)(r * C_ + k0 + kk) * E_ + h * DK_ + d];
    }
    wait_async0();
    __syncthreads();
    v16bf af0 = ld_frag_a(lA, wm);
    v16bf af1 = ld_frag_a(lA, wm + 16);
    for (int tn = 0; tn < 3; ++tn) {
      v16bf bf = ld_frag_b(lB, wn + tn * 16);
      acc[0][tn] = wmma_bf16(af0, bf, acc[0][tn]);
      acc[1][tn] = wmma_bf16(af1, bf, acc[1][tn]);
    }
  }

  const int er = (lane >> 4) << 3, ec = lane & 15;
  for (int tm = 0; tm < 2; ++tm)
    for (int tn = 0; tn < 3; ++tn)
      for (int i = 0; i < 8; ++i) {
        int ii = i0 + wm + tm * 16 + er + i;
        int d  = wn + tn * 16 + ec;
        O[(size_t)(r * C_ + ii) * E_ + h * DK_ + d] = f2bf(acc[tm][tn][i]);
      }
}

// ---------------- col-attention logits: per (h,c) 64x64x96 GEMM, 4 waves, async DB ----
__global__ __launch_bounds__(128) void k_col_logits(const unsigned short* __restrict__ Q,
                                                    const unsigned short* __restrict__ Km,
                                                    float* __restrict__ CP) {
  __shared__ unsigned short lA[2][64 * LDT];
  __shared__ unsigned short lB[2][64 * LDT];
  const int tid = threadIdx.x, lane = tid & 31, wid = tid >> 5;
  const int c = blockIdx.x, h = blockIdx.y;
  const int wm = wid * 16;
  const int srow = tid >> 1, skp = (tid & 1) * 16;

  v8f acc[4];
  const v8f zero = {0.f,0.f,0.f,0.f,0.f,0.f,0.f,0.f};
  for (int j = 0; j < 4; ++j) acc[j] = zero;

  const size_t base = (size_t)(srow * C_ + c) * E_ + h * DK_ + skp;
  unsigned la[2], lb[2];
  la[0] = lds_addr(&lA[0][srow * LDT + skp]);
  la[1] = lds_addr(&lA[1][srow * LDT + skp]);
  lb[0] = lds_addr(&lB[0][srow * LDT + skp]);
  lb[1] = lds_addr(&lB[1][srow * LDT + skp]);

  async_ld_32B(la[0], Q + base);
  async_ld_32B(lb[0], Km + base);

  const int nk = DK_ / 32;   // 3
  for (int t = 0; t < nk; ++t) {
    wait_async0();
    __syncthreads();
    const int p = t & 1;
    if (t + 1 < nk) {
      const int q = p ^ 1;
      const size_t ko = (size_t)(t + 1) * 32;
      async_ld_32B(la[q], Q + base + ko);
      async_ld_32B(lb[q], Km + base + ko);
    }
    v16bf af = ld_frag_a(lA[p], wm);
    for (int tn = 0; tn < 4; ++tn) {
      v16bf bf = ld_frag_b(lB[p], tn * 16);
      acc[tn] = wmma_bf16(af, bf, acc[tn]);
    }
  }

  const int er = (lane >> 4) << 3, ec = lane & 15;
  const size_t obase = (size_t)(h * C_ + c) * R_ * R_;
  for (int tn = 0; tn < 4; ++tn)
    for (int i = 0; i < 8; ++i) {
      int ii = wm + er + i, j = tn * 16 + ec;
      CP[obase + (size_t)ii * R_ + j] = acc[tn][i];
    }
}

// ---------------- col softmax over 64, one wave per row, in place + bf16 copy ----
__global__ __launch_bounds__(256) void k_col_softmax(float* __restrict__ CP,
                                                     unsigned short* __restrict__ CPB) {
  const int tid = threadIdx.x, lane = tid & 31, wid = tid >> 5;
  const size_t row = (size_t)blockIdx.x * 8 + wid;
  float* p = CP + row * 64;
  float a = p[lane], b = p[lane + 32];
  float mx = fmaxf(a, b);
  for (int s = 16; s > 0; s >>= 1) mx = fmaxf(mx, __shfl_xor(mx, s, 32));
  float ea = __expf(a - mx), eb = __expf(b - mx);
  float sum = ea + eb;
  for (int s = 16; s > 0; s >>= 1) sum += __shfl_xor(sum, s, 32);
  const float inv = 1.0f / sum;
  ea *= inv; eb *= inv;
  p[lane] = ea; p[lane + 32] = eb;
  CPB[row * 64 + lane]      = f2bf(ea);
  CPB[row * 64 + lane + 32] = f2bf(eb);
}

// ---------------- col PV: cc[i,c,h,d] = sum_j P[h,c,i,j] v[j,c,h,d]
// per (h,c): 64(i) x 96(d) x 64(j), 4 waves, each 16x96 (6 frags)
__global__ __launch_bounds__(128) void k_col_pv(const unsigned short* __restrict__ P,
                                                const unsigned short* __restrict__ V,
                                                unsigned short* __restrict__ O) {
  __shared__ unsigned short lA[64 * LDT];
  __shared__ unsigned short lB[96 * LDT];
  const int tid = threadIdx.x, lane = tid & 31, wid = tid >> 5;
  const int c = blockIdx.x, h = blockIdx.y;
  const int wm = wid * 16;
  const int srow = tid >> 1, skp = (tid & 1) * 16;
  const unsigned laddr = lds_addr(&lA[srow * LDT + skp]);

  v8f acc[6];
  const v8f zero = {0.f,0.f,0.f,0.f,0.f,0.f,0.f,0.f};
  for (int j = 0; j < 6; ++j) acc[j] = zero;

  for (int k0 = 0; k0 < R_; k0 += 32) {
    __syncthreads();
    const size_t pa = ((size_t)(h * C_ + c) * R_ + srow) * R_ + k0 + skp;
    async_ld_32B(laddr, P + pa);     // A tile via async copy
    for (int it = 0; it < 24; ++it) {               // B tile 32(k=j) x 96(n=d) -> [n][k]
      int idx = tid * 24 + it;
      int kk = idx / 96, d = idx % 96;
      lB[d * LDT + kk] = V[(size_t)((k0 + kk) * C_ + c) * E_ + h * DK_ + d];
    }
    wait_async0();
    __syncthreads();
    v16bf af = ld_frag_a(lA, wm);
    for (int tn = 0; tn < 6; ++tn) {
      v16bf bf = ld_frag_b(lB, tn * 16);
      acc[tn] = wmma_bf16(af, bf, acc[tn]);
    }
  }

  const int er = (lane >> 4) << 3, ec = lane & 15;
  for (int tn = 0; tn < 6; ++tn)
    for (int i = 0; i < 8; ++i) {
      int ii = wm + er + i, d = tn * 16 + ec;
      O[(size_t)(ii * C_ + c) * E_ + h * DK_ + d] = f2bf(acc[tn][i]);
    }
}

// ======================= host launcher =======================
extern "C" void kernel_launch(void* const* d_in, const int* in_sizes, int n_in,
                              void* d_out, int out_size, void* d_ws, size_t ws_size,
                              hipStream_t stream) {
  (void)in_sizes; (void)n_in; (void)out_size; (void)ws_size;

  const float* x    = (const float*)d_in[0];
  const float* lnrw = (const float*)d_in[1];  const float* lnrb = (const float*)d_in[2];
  const float* rqw  = (const float*)d_in[3];  const float* rqb  = (const float*)d_in[4];
  const float* rkw  = (const float*)d_in[5];  const float* rkb  = (const float*)d_in[6];
  const float* rvw  = (const float*)d_in[7];  const float* rvb  = (const float*)d_in[8];
  const float* row_  = (const float*)d_in[9]; const float* rob  = (const float*)d_in[10];
  const float* lncw = (const float*)d_in[11]; const float* lncb = (const float*)d_in[12];
  const float* cqw  = (const float*)d_in[13]; const float* cqb  = (const float*)d_in[14];
  const float* ckw  = (const float*)d_in[15]; const float* ckb  = (const float*)d_in[16];
  const float* cvw  = (const float*)d_in[17]; const float* cvb  = (const float*)d_in[18];
  const float* cow  = (const float*)d_in[19]; const float* cob  = (const float*)d_in[20];
  const float* lnfw = (const float*)d_in[21]; const float* lnfb = (const float*)d_in[22];
  const float* f1w  = (const float*)d_in[23]; const float* f1b  = (const float*)d_in[24];
  const float* f2w  = (const float*)d_in[25]; const float* f2b  = (const float*)d_in[26];

  float* outx = (float*)d_out;                          // [M, E] f32 (final x)
  float* rowp = outx + (size_t)M_ * E_;                 // [H, C, C] f32
  float* colp = rowp + (size_t)H_ * C_ * C_;            // [H, C, R, R] f32

  char* ws = (char*)d_ws;
  size_t cur = 0;
  auto alloc = [&](size_t bytes) -> char* {
    char* p = ws + cur;
    cur += (bytes + 255) & ~(size_t)255;
    return p;
  };
  unsigned short* WQ  = (unsigned short*)alloc((size_t)E_ * E_ * 2);
  unsigned short* WK  = (unsigned short*)alloc((size_t)E_ * E_ * 2);
  unsigned short* WV  = (unsigned short*)alloc((size_t)E_ * E_ * 2);
  unsigned short* WO  = (unsigned short*)alloc((size_t)E_ * E_ * 2);
  unsigned short* WCQ = (unsigned short*)alloc((size_t)E_ * E_ * 2);
  unsigned short* WCK = (unsigned short*)alloc((size_t)E_ * E_ * 2);
  unsigned short* WCV = (unsigned short*)alloc((size_t)E_ * E_ * 2);
  unsigned short* WCO = (unsigned short*)alloc((size_t)E_ * E_ * 2);
  unsigned short* WF1 = (unsigned short*)alloc((size_t)F_ * E_ * 2);
  unsigned short* WF2 = (unsigned short*)alloc((size_t)E_ * F_ * 2);
  unsigned short* XN  = (unsigned short*)alloc((size_t)M_ * E_ * 2);
  unsigned short* QB  = (unsigned short*)alloc((size_t)M_ * E_ * 2);  // also rc/cc
  unsigned short* KB  = (unsigned short*)alloc((size_t)M_ * E_ * 2);  // also FFN h chunk
  unsigned short* VB  = (unsigned short*)alloc((size_t)M_ * E_ * 2);
  float*          RL  = (float*)alloc((size_t)H_ * C_ * C_ * 4);
  unsigned short* RP  = (unsigned short*)alloc((size_t)H_ * C_ * C_ * 2);
  unsigned short* CPB = (unsigned short*)alloc((size_t)H_ * C_ * R_ * R_ * 2);

  auto cast = [&](const float* s, unsigned short* d, size_t n) {
    k_cast<<<dim3((unsigned)((n + 1023) / 1024)), 256, 0, stream>>>(s, d, (int)n);
  };
  cast(rqw, WQ, (size_t)E_ * E_);  cast(rkw, WK, (size_t)E_ * E_);
  cast(rvw, WV, (size_t)E_ * E_);  cast(row_, WO, (size_t)E_ * E_);
  cast(cqw, WCQ, (size_t)E_ * E_); cast(ckw, WCK, (size_t)E_ * E_);
  cast(cvw, WCV, (size_t)E_ * E_); cast(cow, WCO, (size_t)E_ * E_);
  cast(f1w, WF1, (size_t)F_ * E_); cast(f2w, WF2, (size_t)E_ * F_);

  const float srow = 1.0f / (sqrtf(96.0f) * 8.0f);   // dk^-0.5 / sqrt(R)
  const float scol = 1.0f / sqrtf(96.0f);
  const dim3 gE((unsigned)(M_ / 128), (unsigned)(E_ / 128));

  // ---- row attention ----
  k_ln<<<M_, 256, 0, stream>>>(x, lnrw, lnrb, XN);
  k_gemm<<<gE, 256, 0, stream>>>(XN, WQ, rqb, nullptr, nullptr, QB, M_, E_, E_, srow, 0);
  k_gemm<<<gE, 256, 0, stream>>>(XN, WK, rkb, nullptr, nullptr, KB, M_, E_, E_, 1.f, 0);
  k_gemm<<<gE, 256, 0, stream>>>(XN, WV, rvb, nullptr, nullptr, VB, M_, E_, E_, 1.f, 0);
  k_row_logits<<<dim3(2, 2, 8), 256, 0, stream>>>(QB, KB, RL);
  k_row_softmax<<<H_ * C_, 256, 0, stream>>>(RL, rowp, RP);
  k_row_pv<<<dim3(2, R_, H_), 256, 0, stream>>>(RP, VB, QB);
  k_gemm<<<gE, 256, 0, stream>>>(QB, WO, rob, x, outx, nullptr, M_, E_, E_, 1.f, 0);

  // ---- column attention ----
  k_ln<<<M_, 256, 0, stream>>>(outx, lncw, lncb, XN);
  k_gemm<<<gE, 256, 0, stream>>>(XN, WCQ, cqb, nullptr, nullptr, QB, M_, E_, E_, scol, 0);
  k_gemm<<<gE, 256, 0, stream>>>(XN, WCK, ckb, nullptr, nullptr, KB, M_, E_, E_, 1.f, 0);
  k_gemm<<<gE, 256, 0, stream>>>(XN, WCV, cvb, nullptr, nullptr, VB, M_, E_, E_, 1.f, 0);
  k_col_logits<<<dim3(C_, H_), 128, 0, stream>>>(QB, KB, colp);
  k_col_softmax<<<(H_ * C_ * R_) / 8, 256, 0, stream>>>(colp, CPB);
  k_col_pv<<<dim3(C_, H_), 128, 0, stream>>>(CPB, VB, QB);
  k_gemm<<<gE, 256, 0, stream>>>(QB, WCO, cob, outx, outx, nullptr, M_, E_, E_, 1.f, 0);

  // ---- FFN (chunked over tokens; h chunk reuses KB) ----
  k_ln<<<M_, 256, 0, stream>>>(outx, lnfw, lnfb, XN);
  for (int m0 = 0; m0 < M_; m0 += 4096) {
    k_gemm<<<dim3(32, 24), 256, 0, stream>>>(XN + (size_t)m0 * E_, WF1, f1b,
                                             nullptr, nullptr, KB,
                                             4096, F_, E_, 1.f, 1);
    k_gemm<<<dim3(32, 6), 256, 0, stream>>>(KB, WF2, f2b,
                                            outx + (size_t)m0 * E_,
                                            outx + (size_t)m0 * E_, nullptr,
                                            4096, E_, F_, 1.f, 0);
  }
}
